// AdaptiveHarmonicSelector_74706661147116
// MI455X (gfx1250) — matrix operand order repaired and checked by reading, not compiled, not run on hardware
//
#include <hip/hip_runtime.h>
#include <hip/hip_bf16.h>
#include <stdint.h>

#define HH   256
#define DD   768
#define DH   384
#define KSEL 128

typedef float  v8f   __attribute__((ext_vector_type(8)));
typedef __bf16 v8bf  __attribute__((ext_vector_type(8)));
typedef __bf16 v16bf __attribute__((ext_vector_type(16)));
typedef unsigned int u32x4 __attribute__((ext_vector_type(4)));
typedef int    i32x4 __attribute__((ext_vector_type(4)));
typedef int    i32x8 __attribute__((ext_vector_type(8)));

__device__ __forceinline__ float wave_sum_f(float v) {
#pragma unroll
  for (int i = 16; i > 0; i >>= 1) v += __shfl_xor(v, i, 32);
  return v;
}
__device__ __forceinline__ float wave_max_f(float v) {
#pragma unroll
  for (int i = 16; i > 0; i >>= 1) v = fmaxf(v, __shfl_xor(v, i, 32));
  return v;
}
__device__ __forceinline__ unsigned wave_sum_u(unsigned v) {
#pragma unroll
  for (int i = 16; i > 0; i >>= 1) v += __shfl_xor((int)v, i, 32);
  return v;
}

// ---------------------------------------------------------------------------
// Prep: convert f32 weights to bf16 in WMMA-B-fragment tiled layout.
// Tile (ks, nt) = 32(K) x 16(N); 512 bf16 per tile; lane l owns 16 contiguous
// bf16 = K (l<16 ? 0..7,16..23 : 8..15,24..31), N = l%16.
// ---------------------------------------------------------------------------
__global__ void prep_weights(const float* __restrict__ W, __bf16* __restrict__ out,
                             int numN, int ldw, int total) {
  int idx = blockIdx.x * blockDim.x + threadIdx.x;
  if (idx >= total) return;
  int tile = idx >> 9;
  int r = idx & 511;
  int l = r >> 4;
  int j = r & 15;
  int ks = tile / numN;
  int nt = tile - ks * numN;
  int k = ks * 32 + ((j < 8) ? (8 * (l >> 4) + j) : (16 + 8 * (l >> 4) + (j - 8)));
  int n = nt * 16 + (l & 15);
  out[idx] = (__bf16)W[k * ldw + n];
}

// ---------------------------------------------------------------------------
// Kernel 1: h = LN(silu-pre) of (wave_repr @ W1 + b1); SiLU; store bf16 h.
// One block = 16 rows x 384 cols. 8 waves x 3 N-tiles. 24 k-steps of 32.
// ---------------------------------------------------------------------------
__global__ __launch_bounds__(256)
void gemm1_ln_silu(const float* __restrict__ wr, const float* __restrict__ b1,
                   const float* __restrict__ lng, const float* __restrict__ lnb,
                   const __bf16* __restrict__ w1b, __bf16* __restrict__ hout) {
  __shared__ __bf16 shA[16 * 32];
  __shared__ float  shh[16 * DH];
  __shared__ float  shr[2][16][16];

  const int tid  = threadIdx.x;
  const int lane = tid & 31;
  const int w    = tid >> 5;
  const int rowbase = blockIdx.x * 16;

  v8f acc[3] = {};

  const int m   = lane & 15;
  const int sel = lane >> 4;          // 0: K 0..7/16..23, 1: K 8..15/24..31
  const int ar  = tid >> 4;           // staging row 0..15
  const int ac  = (tid & 15) * 2;     // staging col pair
  const float* arow = wr + (size_t)(rowbase + ar) * DD + ac;

  for (int ks = 0; ks < 24; ++ks) {
    __syncthreads();
    float2 f = *(const float2*)(arow + ks * 32);
    shA[ar * 32 + ac]     = (__bf16)f.x;
    shA[ar * 32 + ac + 1] = (__bf16)f.y;
    __syncthreads();

    const __bf16* ap = shA + m * 32 + sel * 8;
    v8bf a0 = *(const v8bf*)(ap);
    v8bf a1 = *(const v8bf*)(ap + 16);
    v16bf av = __builtin_shufflevector(a0, a1, 0,1,2,3,4,5,6,7,8,9,10,11,12,13,14,15);

#pragma unroll
    for (int t = 0; t < 3; ++t) {
      int nt = w * 3 + t;
      const v16bf* bp = (const v16bf*)(w1b + (size_t)((ks * 24 + nt) << 9) + lane * 16);
      acc[t] = __builtin_amdgcn_wmma_f32_16x16x32_bf16(false, av, false, *bp,
                                                       (short)0, acc[t], false, false);
    }
  }
  __syncthreads();

  // scatter C tiles (+b1) into LDS: VGPR r, lane -> row r + 8*sel, col = tile*16 + m
#pragma unroll
  for (int t = 0; t < 3; ++t) {
    int col = (w * 3 + t) * 16 + m;
    float bv = b1[col];
#pragma unroll
    for (int r = 0; r < 8; ++r)
      shh[(r + sel * 8) * DH + col] = acc[t][r] + bv;
  }
  __syncthreads();

  // LayerNorm stats: 16 threads per row
  const int row = tid >> 4;
  const int sub = tid & 15;
  float s = 0.f, s2 = 0.f;
  for (int c = sub; c < DH; c += 16) {
    float v = shh[row * DH + c];
    s += v; s2 += v * v;
  }
  shr[0][row][sub] = s;
  shr[1][row][sub] = s2;
  __syncthreads();
  for (int st = 8; st > 0; st >>= 1) {
    if (sub < st) {
      shr[0][row][sub] += shr[0][row][sub + st];
      shr[1][row][sub] += shr[1][row][sub + st];
    }
    __syncthreads();
  }
  float mu  = shr[0][row][0] * (1.f / DH);
  float var = shr[1][row][0] * (1.f / DH) - mu * mu;
  float rs  = rsqrtf(var + 1e-5f);

  for (int c = sub; c < DH; c += 16) {
    float v = (shh[row * DH + c] - mu) * rs * lng[c] + lnb[c];
    v = v / (1.f + __expf(-v));                        // SiLU
    hout[(size_t)(rowbase + row) * DH + c] = (__bf16)v;
  }
}

// ---------------------------------------------------------------------------
// Kernel 2: scores = h @ W2 + b2; fused gumbel/softmax/top-128 mask; output.
// h tile loaded via Tensor Data Mover into LDS.
// ---------------------------------------------------------------------------
__global__ __launch_bounds__(256)
void gemm2_select(const float* __restrict__ wr, const float* __restrict__ gum,
                  const float* __restrict__ b2, const __bf16* __restrict__ w2b,
                  const __bf16* __restrict__ hsrc, float* __restrict__ out) {
  __shared__ __bf16 shH[16 * DH];     // 12 KB, TDM destination
  __shared__ float  shs[16 * HH];     // 16 KB scores

  const int tid  = threadIdx.x;
  const int lane = tid & 31;
  const int w    = tid >> 5;
  const int rowbase = blockIdx.x * 16;

#if __has_builtin(__builtin_amdgcn_tensor_load_to_lds)
  if (tid < 32) {
    // Tensor DMA descriptor: 2D tile 384(x) x 16(y) of 2-byte elems, row-major.
    unsigned lds_off = (unsigned)(uintptr_t)(&shH[0]);
    unsigned long long ga = (unsigned long long)(uintptr_t)hsrc
                          + (unsigned long long)rowbase * (DH * 2);
    u32x4 g0;
    g0[0] = 1u;                                   // count=1, user descriptor
    g0[1] = lds_off;                              // lds_addr
    g0[2] = (unsigned)(ga & 0xFFFFFFFFull);       // global_addr[31:0]
    g0[3] = (unsigned)((ga >> 32) & 0x1FFFFFFull) // global_addr[56:32]
          | (2u << 30);                           // type=2 (image)
    i32x8 g1;
    g1[0] = (1 << 16);                            // data_size=1 -> 2 bytes
    g1[1] = (int)((DH & 0xFFFF) << 16);           // tensor_dim0 lo16
    g1[2] = (int)((DH >> 16) & 0xFFFF) | (16 << 16); // tensor_dim0 hi | tensor_dim1 lo
    g1[3] = (int)(DH << 16);                      // tensor_dim1 hi | tile_dim0=384
    g1[4] = 16;                                   // tile_dim1=16, tile_dim2=0
    g1[5] = DH;                                   // tensor_dim0_stride lo32
    g1[6] = 0;
    g1[7] = 0;
    i32x4 z4; z4[0] = z4[1] = z4[2] = z4[3] = 0;
#if defined(__clang_major__) && (__clang_major__ >= 23)
    i32x8 z8; z8[0]=z8[1]=z8[2]=z8[3]=z8[4]=z8[5]=z8[6]=z8[7]=0;
    __builtin_amdgcn_tensor_load_to_lds(g0, g1, z4, z4, z8, 0);
#else
    __builtin_amdgcn_tensor_load_to_lds(g0, g1, z4, z4, 0);
#endif
#if __has_builtin(__builtin_amdgcn_s_wait_tensorcnt)
    __builtin_amdgcn_s_wait_tensorcnt(0);
#else
    asm volatile("s_wait_tensorcnt 0x0" ::: "memory");
#endif
  }
#else
  {
    const v8bf* src = (const v8bf*)(hsrc + (size_t)rowbase * DH);
    v8bf* dst = (v8bf*)shH;
    for (int i = tid; i < (16 * DH / 8); i += 256) dst[i] = src[i];
  }
#endif
  __syncthreads();

  v8f acc[2] = {};
  const int m   = lane & 15;
  const int sel = lane >> 4;

  for (int ks = 0; ks < 12; ++ks) {
    const __bf16* ap = shH + m * DH + ks * 32 + sel * 8;
    v8bf a0 = *(const v8bf*)(ap);
    v8bf a1 = *(const v8bf*)(ap + 16);
    v16bf av = __builtin_shufflevector(a0, a1, 0,1,2,3,4,5,6,7,8,9,10,11,12,13,14,15);
#pragma unroll
    for (int t = 0; t < 2; ++t) {
      int nt = w * 2 + t;
      const v16bf* bp = (const v16bf*)(w2b + (size_t)((ks * 16 + nt) << 9) + lane * 16);
      acc[t] = __builtin_amdgcn_wmma_f32_16x16x32_bf16(false, av, false, *bp,
                                                       (short)0, acc[t], false, false);
    }
  }

#pragma unroll
  for (int t = 0; t < 2; ++t) {
    int col = (w * 2 + t) * 16 + m;
    float bv = b2[col];
#pragma unroll
    for (int r = 0; r < 8; ++r)
      shs[(r + sel * 8) * HH + col] = acc[t][r] + bv;
  }
  __syncthreads();

  // Epilogue: each wave handles rows w and w+8; lane owns elems e = c*32+lane.
  for (int rr = 0; rr < 2; ++rr) {
    const int row = w + rr * 8;
    const size_t R = (size_t)(rowbase + row);
    float sc[8], fr[8], am[8], ph[8], gu[8];
#pragma unroll
    for (int c = 0; c < 8; ++c) {
      int e = c * 32 + lane;
      sc[c] = shs[row * HH + e];
      fr[c] = wr[R * DD + e];
      am[c] = wr[R * DD + 256 + e];
      ph[c] = wr[R * DD + 512 + e];
      gu[c] = gum[R * HH + e];
    }
    float asum = 0.f;
#pragma unroll
    for (int c = 0; c < 8; ++c) asum += am[c];
    asum = wave_sum_f(asum);
    float inva = 1.f / (asum + 1e-8f);

    float p[8], mx = -3.4e38f;
#pragma unroll
    for (int c = 0; c < 8; ++c) {
      float g = -__logf(-__logf(gu[c] + 1e-8f) + 1e-8f);
      p[c] = sc[c] + am[c] * inva + g;          // TEMP = 1
      mx = fmaxf(mx, p[c]);
    }
    mx = wave_max_f(mx);
    float ex[8], es = 0.f;
#pragma unroll
    for (int c = 0; c < 8; ++c) { ex[c] = __expf(p[c] - mx); es += ex[c]; }
    es = wave_sum_f(es);
    float invs = 1.f / es;

    // order-preserving uint keys
    unsigned key[8];
#pragma unroll
    for (int c = 0; c < 8; ++c) {
      unsigned u = __float_as_uint(p[c]);
      key[c] = (u & 0x80000000u) ? ~u : (u | 0x80000000u);
    }
    // binary search largest T with count(key >= T) >= KSEL
    unsigned lo = 0u, hi = 0xFFFFFFFFu;
    while (lo < hi) {
      unsigned mid = lo + ((hi - lo) >> 1) + 1u;
      unsigned cnt = 0;
#pragma unroll
      for (int c = 0; c < 8; ++c) cnt += (key[c] >= mid) ? 1u : 0u;
      cnt = wave_sum_u(cnt);
      if (cnt >= (unsigned)KSEL) lo = mid; else hi = mid - 1u;
    }
    const unsigned T = lo;
    unsigned cg = 0;
#pragma unroll
    for (int c = 0; c < 8; ++c) cg += (key[c] > T) ? 1u : 0u;
    cg = wave_sum_u(cg);
    int sel_eq = KSEL - (int)cg;                 // #ties to keep (lowest index first)

    int base = 0;
    const unsigned lt_mask = (lane == 0) ? 0u : (0xFFFFFFFFu >> (32 - lane));
    float mk[8];
#pragma unroll
    for (int c = 0; c < 8; ++c) {
      bool eq = (key[c] == T);
      unsigned bal = (unsigned)__ballot(eq);
      int pre = __popc(bal & lt_mask);
      bool inc = eq && ((base + pre) < sel_eq);
      base += __popc(bal);
      float hard = ((key[c] > T) || inc) ? 1.f : 0.f;
      float soft = ex[c] * invs;
      mk[c] = (hard + soft) - soft;              // straight-through value
    }
#pragma unroll
    for (int c = 0; c < 8; ++c) {
      int e = c * 32 + lane;
      out[R * DD + e]       = fr[c] * mk[c];
      out[R * DD + 256 + e] = am[c] * mk[c];
      out[R * DD + 512 + e] = ph[c] * mk[c];
    }
  }
}

// ---------------------------------------------------------------------------
extern "C" void kernel_launch(void* const* d_in, const int* in_sizes, int n_in,
                              void* d_out, int out_size, void* d_ws, size_t ws_size,
                              hipStream_t stream) {
  (void)in_sizes; (void)n_in; (void)out_size; (void)ws_size;
  const float* wr  = (const float*)d_in[0];
  const float* gum = (const float*)d_in[1];
  const float* W1  = (const float*)d_in[2];
  const float* b1  = (const float*)d_in[3];
  const float* lng = (const float*)d_in[4];
  const float* lnb = (const float*)d_in[5];
  const float* W2  = (const float*)d_in[6];
  const float* b2  = (const float*)d_in[7];
  float* out = (float*)d_out;

  char* ws = (char*)d_ws;
  __bf16* w1b  = (__bf16*)(ws);                 // 24*24*512*2 = 589824 B
  __bf16* w2b  = (__bf16*)(ws + 589824);        // 12*16*512*2 = 196608 B
  __bf16* hbuf = (__bf16*)(ws + 786432);        // 32768*384*2 = 25.2 MB

  const int t1 = 24 * 24 * 512;
  const int t2 = 12 * 16 * 512;
  prep_weights<<<(t1 + 255) / 256, 256, 0, stream>>>(W1, w1b, 24, DH, t1);
  prep_weights<<<(t2 + 255) / 256, 256, 0, stream>>>(W2, w2b, 16, HH, t2);

  const int mtiles = (8 * 4096) / 16;           // 2048
  gemm1_ln_silu<<<mtiles, 256, 0, stream>>>(wr, b1, lng, lnb, w1b, hbuf);
  gemm2_select<<<mtiles, 256, 0, stream>>>(wr, gum, b2, w2b, hbuf, out);
}
